// MOELayer_20444044329142
// MI455X (gfx1250) — matrix operand order repaired and checked by reading, not compiled, not run on hardware
//
#include <hip/hip_runtime.h>
#include <hip/hip_bf16.h>

typedef __attribute__((ext_vector_type(16))) __bf16 v16bf;
typedef __attribute__((ext_vector_type(8)))  float  v8f;

#define NTOK  4096   // B*T
#define CDIM  1024
#define HDIM  4096
#define NEXP  8
#define NSLOT 9      // 8 routed + 1 shared

// f32 -> bf16 via native conversion (RNE on gfx1250)
__device__ __forceinline__ unsigned short f2bf(float x) {
  __bf16 b = (__bf16)x;
  return __builtin_bit_cast(unsigned short, b);
}
__device__ __forceinline__ unsigned pack2_bf16(float lo, float hi) {
  return (unsigned)f2bf(lo) | ((unsigned)f2bf(hi) << 16);
}

// --- CDNA5 async global->LDS copy (ASYNCcnt path), 16B per lane -------------
__device__ __forceinline__ void async_copy_b128(unsigned lds_off, const void* gptr) {
  asm volatile("global_load_async_to_lds_b128 %0, %1, off"
               :: "v"(lds_off), "v"(gptr) : "memory");
}
__device__ __forceinline__ void wait_asynccnt0() {
#if __has_builtin(__builtin_amdgcn_s_wait_asynccnt)
  __builtin_amdgcn_s_wait_asynccnt(0);
#else
  asm volatile("s_wait_asynccnt 0x0" ::: "memory");
#endif
}
__device__ __forceinline__ unsigned lds_offset_of(const void* p) {
  return (unsigned)(unsigned long long)p;  // LDS aperture: low 32 bits = LDS byte offset
}

// ---------------------------------------------------------------------------
// Small helper kernels
// ---------------------------------------------------------------------------
__global__ void zero_aux_kernel(float* facc) {
  int i = threadIdx.x;
  if (i < 16) facc[i] = 0.f;   // facc[0..7], Pacc[8..15]
}

__global__ __launch_bounds__(256) void cvt_bf16_kernel(
    const float* __restrict__ in, unsigned short* __restrict__ out, int n) {
  int i = blockIdx.x * blockDim.x + threadIdx.x;
  int stride = gridDim.x * blockDim.x;
  for (; i < n; i += stride) out[i] = f2bf(in[i]);
}

__global__ void finalize_aux_kernel(const float* __restrict__ facc,
                                    float* __restrict__ outp) {
  if (threadIdx.x == 0 && blockIdx.x == 0) {
    const float* P = facc + 8;
    float s = 0.f;
#pragma unroll
    for (int e = 0; e < NEXP; e++) s += facc[e] * P[e];
    *outp = (float)NEXP * s;
  }
}

// ---------------------------------------------------------------------------
// Router: one wave32 per token. logits = X[t,:] @ router_w; softmax; top-2.
// ---------------------------------------------------------------------------
__global__ __launch_bounds__(256) void router_kernel(
    const float* __restrict__ X, const float* __restrict__ RW,
    float* __restrict__ wfull, float* __restrict__ facc) {
  const int lane  = threadIdx.x & 31;
  const int token = blockIdx.x * 8 + (threadIdx.x >> 5);

  float acc[NEXP];
#pragma unroll
  for (int e = 0; e < NEXP; e++) acc[e] = 0.f;

  const float* xr = X + (size_t)token * CDIM;
  for (int c = lane; c < CDIM; c += 32) {
    float xv = xr[c];
    const float* rw = RW + (size_t)c * NEXP;
#pragma unroll
    for (int e = 0; e < NEXP; e++) acc[e] += xv * rw[e];
  }
#pragma unroll
  for (int off = 16; off >= 1; off >>= 1) {
#pragma unroll
    for (int e = 0; e < NEXP; e++) acc[e] += __shfl_xor(acc[e], off, 32);
  }
  float mx = acc[0];
#pragma unroll
  for (int e = 1; e < NEXP; e++) mx = fmaxf(mx, acc[e]);
  float p[NEXP], sum = 0.f;
#pragma unroll
  for (int e = 0; e < NEXP; e++) { p[e] = __expf(acc[e] - mx); sum += p[e]; }
  float inv = 1.f / sum;
#pragma unroll
  for (int e = 0; e < NEXP; e++) p[e] *= inv;
  int i1 = 0;
#pragma unroll
  for (int e = 1; e < NEXP; e++) if (p[e] > p[i1]) i1 = e;
  int i2 = (i1 == 0) ? 1 : 0;
#pragma unroll
  for (int e = 0; e < NEXP; e++) if (e != i1 && p[e] > p[i2]) i2 = e;

  if (lane == 0) {
    float* wrow = wfull + (size_t)token * NSLOT;
#pragma unroll
    for (int e = 0; e < NEXP; e++)
      wrow[e] = (e == i1) ? p[i1] : ((e == i2) ? p[i2] : 0.f);
    wrow[NEXP] = 1.f;  // shared expert slot
    const float invN = 1.f / (float)NTOK;
    float* Pacc = facc + 8;
#pragma unroll
    for (int e = 0; e < NEXP; e++) atomicAdd(&Pacc[e], p[e] * invN);
    atomicAdd(&facc[i1], invN);
    atomicAdd(&facc[i2], invN);
  }
}

// ---------------------------------------------------------------------------
// WMMA GEMM: C[M x Nc] = A(bf16 row-major) @ B(f32->bf16 on load, row-major)
// Block 128(M) x 128(N) x 64(K): 8 waves in a 4(M) x 2(N) grid, wave tile
// 32x64 (2x4 accumulators) -> 16 v_wmma_f32_16x16x32_bf16 per barrier pair.
// A staged with per-lane global_load_async_to_lds_b128 (fragment scatter in
// the per-lane LDS dest address, ASYNCcnt ordered); B staged through VGPRs
// for the f32->bf16 conversion. Fragment layouts per CDNA5 ISA 7.12.2 so a
// wave reads each fragment as one contiguous 32-byte LDS region.
// MODE 0: D = relu(acc + bias) -> bf16   MODE 1: D (+)= w * (acc + bias) f32
// ---------------------------------------------------------------------------
template <int MODE>
__global__ __launch_bounds__(256) void gemm_wmma_kernel(
    const unsigned short* __restrict__ A, const float* __restrict__ Bw,
    const float* __restrict__ bias, void* __restrict__ Cout,
    const float* __restrict__ wfull, int slot, int K, int Ncols, int beta) {
  __shared__ unsigned short lA[2][8][32][16];  // 16 KB: [kstep][Msub][lane][elem]
  __shared__ unsigned short lB[2][8][32][16];  // 16 KB: [kstep][Nsub][lane][elem]

  const int tid  = threadIdx.x;
  const int wid  = tid >> 5, lane = tid & 31;
  const int wm   = wid >> 1, wn = wid & 1;      // 4 x 2 wave grid
  const int blockM = blockIdx.y * 128;
  const int blockN = blockIdx.x * 128;

  v8f acc[2][4] = {};

  // A-loader invariants: thread covers row ar, K-octet pairs {base,base+1} in
  // both 32-wide K-steps (octet o: kstep=o>>2, half=(o&3)&1, eh=(o&3)>>1)
  const int ar = tid >> 1, abase = (tid & 1) * 2;   // row 0..127, oq base 0 or 2
  const int aeh = abase >> 1;                       // 0 or 1
  const unsigned aL00 = lds_offset_of(&lA[0][ar >> 4][(ar & 15) +  0][aeh * 8]); // oq=base   (half 0)
  const unsigned aL01 = lds_offset_of(&lA[0][ar >> 4][(ar & 15) + 16][aeh * 8]); // oq=base+1 (half 1)
  const unsigned aL10 = lds_offset_of(&lA[1][ar >> 4][(ar & 15) +  0][aeh * 8]);
  const unsigned aL11 = lds_offset_of(&lA[1][ar >> 4][(ar & 15) + 16][aeh * 8]);
  const unsigned short* aG = A + (size_t)(blockM + ar) * K + abase * 8;

  // B-loader invariants
  const int bkp = tid >> 4, bng = tid & 15;        // k-pair 0..15, n-group 0..15
  const int bk = bkp * 2, bn0 = bng * 8;
  const int bi = (bk & 7) + 8 * (bk >> 4), bkh = (bk >> 3) & 1;

  for (int k0 = 0; k0 < K; k0 += 64) {
    // --- A tile (128 x 64 bf16): 4 async b128 copies per lane, LDS-permuted
    async_copy_b128(aL00, aG + k0);          // kstep 0, oq=base
    async_copy_b128(aL01, aG + k0 + 8);      // kstep 0, oq=base+1
    async_copy_b128(aL10, aG + k0 + 32);     // kstep 1, oq=base
    async_copy_b128(aL11, aG + k0 + 40);     // kstep 1, oq=base+1
    // --- B tile (64 x 128 f32 -> bf16), fragment scatter
#pragma unroll
    for (int ks = 0; ks < 2; ks++) {
      const float* s0 = Bw + (size_t)(k0 + ks * 32 + bk) * Ncols + blockN + bn0;
      const float* s1 = s0 + Ncols;
      float4 a0 = *(const float4*)s0, a1 = *(const float4*)(s0 + 4);
      float4 b0 = *(const float4*)s1, b1 = *(const float4*)(s1 + 4);
      float r0[8] = {a0.x, a0.y, a0.z, a0.w, a1.x, a1.y, a1.z, a1.w};
      float r1[8] = {b0.x, b0.y, b0.z, b0.w, b1.x, b1.y, b1.z, b1.w};
#pragma unroll
      for (int j = 0; j < 8; j++) {
        const int n = bn0 + j;
        *(unsigned*)&lB[ks][n >> 4][(n & 15) + 16 * bkh][bi] = pack2_bf16(r0[j], r1[j]);
      }
    }
    wait_asynccnt0();   // own async copies landed in LDS
    __syncthreads();    // everyone's A (async) + B (ds) stores visible

#pragma unroll
    for (int ks = 0; ks < 2; ks++) {
      v16bf af[2], bf[4];
#pragma unroll
      for (int mi = 0; mi < 2; mi++)
        af[mi] = *(const v16bf*)&lA[ks][2 * wm + mi][lane][0];
#pragma unroll
      for (int ni = 0; ni < 4; ni++)
        bf[ni] = *(const v16bf*)&lB[ks][4 * wn + ni][lane][0];
#pragma unroll
      for (int mi = 0; mi < 2; mi++)
#pragma unroll
        for (int ni = 0; ni < 4; ni++)
          acc[mi][ni] = __builtin_amdgcn_wmma_f32_16x16x32_bf16(
              false, af[mi], false, bf[ni], (short)0, acc[mi][ni], false, false);
    }
    __syncthreads();
  }

  // --- epilogue: C/D frag layout: row = r + 8*(lane/16), col = lane%16
  const int lhalf = lane >> 4, lcol = lane & 15;
#pragma unroll
  for (int mi = 0; mi < 2; mi++) {
#pragma unroll
    for (int ni = 0; ni < 4; ni++) {
      const int col = blockN + wn * 64 + ni * 16 + lcol;
      const float bv = bias[col];
#pragma unroll
      for (int r = 0; r < 8; r++) {
        const int row = blockM + wm * 32 + mi * 16 + r + 8 * lhalf;
        float v = acc[mi][ni][r] + bv;
        if (MODE == 0) {
          v = v > 0.f ? v : 0.f;
          ((unsigned short*)Cout)[(size_t)row * Ncols + col] = f2bf(v);
        } else {
          const float w = wfull[(size_t)row * NSLOT + slot];
          float* po = (float*)Cout + (size_t)row * Ncols + col;
          const float o = w * v;
          if (beta) *po += o; else *po = o;
        }
      }
    }
  }
}

// ---------------------------------------------------------------------------
// Host launcher
// ---------------------------------------------------------------------------
extern "C" void kernel_launch(void* const* d_in, const int* in_sizes, int n_in,
                              void* d_out, int out_size, void* d_ws, size_t ws_size,
                              hipStream_t stream) {
  const float* X        = (const float*)d_in[0];
  const float* router_w = (const float*)d_in[1];
  const float* sh_w1    = (const float*)d_in[2];
  const float* sh_b1    = (const float*)d_in[3];
  const float* sh_w2    = (const float*)d_in[4];
  const float* sh_b2    = (const float*)d_in[5];
  const float* rt_w1    = (const float*)d_in[6];
  const float* rt_b1    = (const float*)d_in[7];
  const float* rt_w2    = (const float*)d_in[8];
  const float* rt_b2    = (const float*)d_in[9];
  float* out = (float*)d_out;

  // workspace layout
  char* ws = (char*)d_ws;
  unsigned short* Xbf   = (unsigned short*)ws;                                  // 8 MB
  unsigned short* H1    = (unsigned short*)(ws + (size_t)8 * 1024 * 1024);      // 32 MB
  float*          wfull = (float*)(ws + (size_t)40 * 1024 * 1024);              // 144 KB
  float*          facc  = (float*)(ws + (size_t)40 * 1024 * 1024 + 256 * 1024); // 16 floats

  zero_aux_kernel<<<1, 32, 0, stream>>>(facc);
  cvt_bf16_kernel<<<2048, 256, 0, stream>>>(X, Xbf, NTOK * CDIM);
  router_kernel<<<NTOK / 8, 256, 0, stream>>>(X, router_w, wfull, facc);

  for (int s = 0; s < NSLOT; s++) {
    const float* W1 = (s < NEXP) ? rt_w1 + (size_t)s * CDIM * HDIM : sh_w1;
    const float* b1 = (s < NEXP) ? rt_b1 + (size_t)s * HDIM        : sh_b1;
    const float* W2 = (s < NEXP) ? rt_w2 + (size_t)s * HDIM * CDIM : sh_w2;
    const float* b2 = (s < NEXP) ? rt_b2 + (size_t)s * CDIM        : sh_b2;

    gemm_wmma_kernel<0><<<dim3(HDIM / 128, NTOK / 128), 256, 0, stream>>>(
        Xbf, W1, b1, (void*)H1, nullptr, s, CDIM, HDIM, 0);
    gemm_wmma_kernel<1><<<dim3(CDIM / 128, NTOK / 128), 256, 0, stream>>>(
        H1, W2, b2, (void*)out, wfull, s, HDIM, CDIM, (s > 0) ? 1 : 0);
  }

  finalize_aux_kernel<<<1, 1, 0, stream>>>(facc, out + (size_t)NTOK * CDIM);
}
